// MetaKernel_26259430047925
// MI455X (gfx1250) — compile-verified, matrix operand order checked by reading, not run on hardware
//
#include <hip/hip_runtime.h>
#include <hip/hip_bf16.h>

// ---------------------------------------------------------------------------
// MI455X (gfx1250) fused kernel for the windowed positional-MLP convolution.
// Both GEMMs (h@W2 and xp@Wc^T) run on v_wmma_f32_16x16x32_f16.
// Round 2: branch-free OOB handling, x-window staged in LDS (coalesced),
// constants staged in LDS, deeper unroll on the main WMMA loop.
// ---------------------------------------------------------------------------

typedef __attribute__((ext_vector_type(16))) _Float16 v16h;
typedef __attribute__((ext_vector_type(8)))  float    v8f;
typedef __attribute__((ext_vector_type(4)))  unsigned int v4u;

union Frag16 { v16h h; v4u u[2]; };
union Half8  { _Float16 h[8]; v4u u; };

#define KS   4
#define ST   2
#define CI   64
#define CO   128
#define HH_  64
#define WW_  1024
#define HO   32
#define WO   512
#define NPOS (4 * HO * WO)          // 65536 positions
#define AZI_F 0.006135923151542565f // 2*pi/1024
#define INC_F 0.0073f

// x window per WG: 64 ch x 4 rows x 34 cols, col-padded to 35 (bank spread)
#define XW_COLS 34
#define XW_PITCH 35

// ---------------------------------------------------------------------------
// prep: Wc -> f16 (row-major, k contiguous), W2^T -> f16, G[tap][c] table
// ---------------------------------------------------------------------------
__global__ __launch_bounds__(256) void prep_kernel(
    const float* __restrict__ W1, const float* __restrict__ W2,
    const float* __restrict__ Wc,
    _Float16* __restrict__ Wc_h, _Float16* __restrict__ W2t_h,
    float* __restrict__ G)
{
    int idx = blockIdx.x * 256 + threadIdx.x;
    if (idx < CO * CI * KS * KS)           // 131072
        Wc_h[idx] = (_Float16)Wc[idx];

    if (blockIdx.x == 0) {
        for (int i = threadIdx.x; i < CI * CI; i += 256) {
            int n = i >> 6, c = i & 63;
            W2t_h[n * CI + c] = (_Float16)W2[c * CI + n];   // W2t[n][c] = W2[c][n]
        }
        for (int i = threadIdx.x; i < 16 * CI; i += 256) {
            int tap = i >> 6, c = i & 63;
            float da = (float)((tap >> 2) - 2);   // inc direction
            float db = (float)((tap & 3) - 2);    // azi direction
            float cA = cosf(AZI_F * db), sA = sinf(AZI_F * db);
            float cI = cosf(INC_F * da), sI = sinf(INC_F * da);
            G[i] = cA * cI * W1[c] + cA * sI * W1[CI + c] + sA * W1[2 * CI + c];
        }
    }
}

// ---------------------------------------------------------------------------
// fused: per WG: 16 consecutive output positions (same b,ho; wo0..wo0+15)
// ---------------------------------------------------------------------------
__global__ __launch_bounds__(256) void fused_kernel(
    const float* __restrict__ x,  const float* __restrict__ r,
    const float* __restrict__ W1, const float* __restrict__ b1,
    const float* __restrict__ b2, const float* __restrict__ bc,
    const _Float16* __restrict__ Wc_h, const _Float16* __restrict__ W2t_h,
    const float* __restrict__ G,
    float* __restrict__ out, float* __restrict__ rcen)
{
    __shared__ float    rp_s[16 * 16];                 //  1 KB  r window
    __shared__ float    g_s[16 * CI];                  //  4 KB  G table
    __shared__ float    w1_s[CI];                      // W1 row 0
    __shared__ float    b1_s[CI];
    __shared__ float    xw_s[CI * 4 * XW_PITCH];       // 35 KB  x window (0-padded)
    __shared__ _Float16 h_s[256 * CI];                 // 32 KB  hidden acts (f16)
    __shared__ _Float16 w2t_s[CI * CI];                //  8 KB  W2^T (f16)
    __shared__ _Float16 A_s[16 * CI * 16];             // 32 KB  xp tile [pos][k]

    const int tid  = threadIdx.x;
    const int lane = tid & 31;                // wave32 on gfx1250
    const int wave = tid >> 5;
    const int pos0 = blockIdx.x * 16;
    const int bb   = pos0 >> 14;              // / (HO*WO)
    const int ho   = (pos0 >> 9) & 31;
    const int wo0  = pos0 & 511;
    const int wwbase = 2 * wo0 - 1;

    // --- stage W2^T into LDS (512 x 16B) ---
    {
        const v4u* src = (const v4u*)W2t_h;
        v4u*       dst = (v4u*)w2t_s;
        dst[tid]       = src[tid];
        dst[tid + 256] = src[tid + 256];
    }
    // --- stage G / W1 row0 / b1 ---
    for (int i = tid; i < 16 * CI; i += 256) g_s[i] = G[i];
    if (tid < CI) { w1_s[tid] = W1[tid]; b1_s[tid] = b1[tid]; }

    // --- phase 1a: r window (wrap in H, const 100.0 in W), r_center ---
    {
        int p = tid >> 4, tap = tid & 15;
        int a = tap >> 2, bk = tap & 3;       // rp[...,a,bk] = rpad[2ho+bk, 2wo+a]
        int hh = (2 * ho + bk - 1) & 63;      // wrap pad in H
        int ww = 2 * (wo0 + p) + a - 1;       // const pad in W
        int wc = min(max(ww, 0), WW_ - 1);    // branch-free clamp + select
        float rl = r[(bb * HH_ + hh) * WW_ + wc];
        float rv = (ww == wc) ? rl : 100.0f;
        rp_s[tid] = rv;
        if (tap == 10) rcen[pos0 + p] = rv;   // center tap (a=2,b=2)
    }

    // --- phase 1b: x window -> LDS, coalesced, OOB zeroed at fill time ---
    for (int i = tid; i < CI * 4 * XW_COLS; i += 256) {
        int c   = i / (4 * XW_COLS);
        int rem = i - c * (4 * XW_COLS);
        int row = rem / XW_COLS;              // = bk tap component
        int col = rem - row * XW_COLS;        // = 2*p + a
        int hh  = (2 * ho - 1 + row) & 63;    // wrap pad in H
        int ww  = wwbase + col;               // zero pad in W
        int wc  = min(max(ww, 0), WW_ - 1);
        float xv = x[((bb * CI + c) * HH_ + hh) * WW_ + wc];
        xw_s[(c * 4 + row) * XW_PITCH + col] = (ww == wc) ? xv : 0.0f;
    }
    __syncthreads();

    // --- phase 1c: h[p][tap][c] = leaky(rp*G - rc*W1[0,c] + b1[c]) ---
    for (int i = tid; i < 16 * 16 * CI; i += 256) {
        int p   = i >> 10;
        int tap = (i >> 6) & 15;
        int c   = i & 63;
        float rc = rp_s[p * 16 + 10];
        float v  = rp_s[p * 16 + tap] * g_s[tap * CI + c] - rc * w1_s[c] + b1_s[c];
        v = (v > 0.0f) ? v : 0.2f * v;        // leaky_relu(0.2)
        h_s[(p * 16 + tap) * CI + c] = (_Float16)v;
    }
    __syncthreads();

    const int rlo   = lane & 15;
    const int khalf = (lane < 16) ? 0 : 8;    // A/B 16-bit fragment K split
    const int mhalf = (lane & 16) ? 8 : 0;    // C/D row split

    // hoist per-lane b2 values (4 N-tiles per lane in phase 2)
    float b2v4[4];
    #pragma unroll
    for (int nt = 0; nt < 4; ++nt) b2v4[nt] = b2[nt * 16 + rlo];

    // --- phase 2: w = h@W2 + b2 via WMMA, fuse x-window product -> A tile ---
    for (int j = 0; j < 8; ++j) {
        int tileid = wave * 8 + j;            // 64 tiles = 16 pos x 4 ntiles
        int p  = tileid >> 2;
        int nt = tileid & 3;
        int cch = nt * 16 + rlo;              // MLP output channel = x channel
        v8f acc = {};
        #pragma unroll
        for (int ks = 0; ks < 2; ++ks) {      // K=64 -> 2 x k32
            int koff = ks * 32 + khalf;
            Frag16 af, bf;
            const _Float16* ap = &h_s[(p * 16 + rlo) * CI + koff];
            af.u[0] = *(const v4u*)(ap);
            af.u[1] = *(const v4u*)(ap + 16);
            const _Float16* bp = &w2t_s[cch * CI + koff];
            bf.u[0] = *(const v4u*)(bp);
            bf.u[1] = *(const v4u*)(bp + 16);
            acc = __builtin_amdgcn_wmma_f32_16x16x32_f16(
                false, af.h, false, bf.h, (short)0, acc, false, false);
        }
        float b2v = b2v4[nt];
        const float* xrow = &xw_s[cch * (4 * XW_PITCH)];
        Half8 pk;
        #pragma unroll
        for (int i2 = 0; i2 < 8; ++i2) {
            int m  = i2 + mhalf;              // tap index (D-row)
            int a  = m >> 2, bk = m & 3;
            float xu = xrow[bk * XW_PITCH + 2 * p + a];   // no predication
            pk.h[i2] = (_Float16)(xu * (acc[i2] + b2v));
        }
        // k = cch*16 + tap, 8 consecutive taps -> one b128 LDS store
        *(v4u*)(&A_s[p * 1024 + cch * 16 + mhalf]) = pk.u;
    }
    __syncthreads();

    // --- phase 3: out[16 pos][128 out] = A(16x1024) @ Wc^T via WMMA ---
    {
        int o = wave * 16 + rlo;              // one N-tile per wave
        v8f acc = {};
        const _Float16* brow = Wc_h + o * 1024;
        const _Float16* arow = &A_s[rlo * 1024];
        #pragma unroll 8
        for (int kk = 0; kk < 32; ++kk) {
            int koff = kk * 32 + khalf;
            Frag16 af, bf;
            const _Float16* ap = arow + koff;
            af.u[0] = *(const v4u*)(ap);
            af.u[1] = *(const v4u*)(ap + 16);
            const _Float16* bp = brow + koff;
            __builtin_prefetch(bp + 64, 0, 0);    // global_prefetch next k-chunk
            bf.u[0] = *(const v4u*)(bp);
            bf.u[1] = *(const v4u*)(bp + 16);
            acc = __builtin_amdgcn_wmma_f32_16x16x32_f16(
                false, af.h, false, bf.h, (short)0, acc, false, false);
        }
        float bcv = bc[o];
        // D rows = consecutive wo -> contiguous stores
        float* obase = out + (((bb * CO + o) * HO + ho) * WO) + wo0 + mhalf;
        float4 lo = make_float4(acc[0] + bcv, acc[1] + bcv, acc[2] + bcv, acc[3] + bcv);
        float4 hi = make_float4(acc[4] + bcv, acc[5] + bcv, acc[6] + bcv, acc[7] + bcv);
        *(float4*)(obase)     = lo;
        *(float4*)(obase + 4) = hi;
    }
}

// ---------------------------------------------------------------------------
extern "C" void kernel_launch(void* const* d_in, const int* in_sizes, int n_in,
                              void* d_out, int out_size, void* d_ws, size_t ws_size,
                              hipStream_t stream) {
    const float* x  = (const float*)d_in[0];
    const float* r  = (const float*)d_in[1];
    const float* W1 = (const float*)d_in[2];
    const float* b1 = (const float*)d_in[3];
    const float* W2 = (const float*)d_in[4];
    const float* b2 = (const float*)d_in[5];
    const float* Wc = (const float*)d_in[6];
    const float* bc = (const float*)d_in[7];
    float* out = (float*)d_out;

    char* ws = (char*)d_ws;
    _Float16* Wc_h  = (_Float16*)(ws);                    // 262144 B
    _Float16* W2t_h = (_Float16*)(ws + 262144);           //   8192 B
    float*    G     = (float*)   (ws + 262144 + 8192);    //   4096 B

    float* rcen = out + 4 * CO * HO * WO;                 // tuple tail: r_center

    prep_kernel<<<512, 256, 0, stream>>>(W1, W2, Wc, Wc_h, W2t_h, G);
    fused_kernel<<<NPOS / 16, 256, 0, stream>>>(x, r, W1, b1, b2, bc,
                                                Wc_h, W2t_h, G, out, rcen);
}
